// Top1MoE_50946902065583
// MI455X (gfx1250) — matrix operand order repaired and checked by reading, not compiled, not run on hardware
//
#include <hip/hip_runtime.h>

#define TOK  4096      // B*S
#define DIM  1024      // D
#define NEXP 8         // E
#define MTILES (TOK / 16)

typedef __attribute__((ext_vector_type(16))) __bf16 v16bf;
typedef __attribute__((ext_vector_type(8)))  __bf16 v8bf;
typedef __attribute__((ext_vector_type(8)))  float  v8f;
typedef int v4i32 __attribute__((vector_size(16)));   // b128 payload type

static __device__ inline unsigned pack2bf(float a, float b) {
    unsigned short ha = __builtin_bit_cast(unsigned short, (__bf16)a);
    unsigned short hb = __builtin_bit_cast(unsigned short, (__bf16)b);
    return ((unsigned)hb << 16) | (unsigned)ha;
}

// ---------------------------------------------------------------------------
// Kernel 0: zero per-expert counters (graph-safe, redone every call).
// ---------------------------------------------------------------------------
__global__ void moe_init_kernel(int* __restrict__ counts) {
    if (threadIdx.x < NEXP) counts[threadIdx.x] = 0;
}

// ---------------------------------------------------------------------------
// Kernel 1: fp32 -> bf16 bulk convert (8 floats -> one 16B store per thread).
// ---------------------------------------------------------------------------
__global__ __launch_bounds__(256) void cvt_bf16_kernel(
    const float* __restrict__ src, __bf16* __restrict__ dst, int n8) {
    int i = blockIdx.x * 256 + threadIdx.x;
    if (i >= n8) return;
    const float4* s = (const float4*)src + (size_t)i * 2;
    float4 q0 = s[0];
    float4 q1 = s[1];
    uint4 o;
    o.x = pack2bf(q0.x, q0.y);
    o.y = pack2bf(q0.z, q0.w);
    o.z = pack2bf(q1.x, q1.y);
    o.w = pack2bf(q1.z, q1.w);
    ((uint4*)dst)[i] = o;
}

// ---------------------------------------------------------------------------
// Kernel 2: router. One wave per token: 8 logits, softmax top-1 weight,
// append token to its expert's list.
// ---------------------------------------------------------------------------
__global__ __launch_bounds__(256) void moe_router_kernel(
    const float* __restrict__ x, const float* __restrict__ rw,
    const float* __restrict__ rb, float* __restrict__ top1w,
    int* __restrict__ counts, int* __restrict__ perm) {
    int lane  = threadIdx.x & 31;
    int wave  = threadIdx.x >> 5;
    int token = blockIdx.x * 8 + wave;
    if (token >= TOK) return;

    const float* xr = x + (size_t)token * DIM;
    float lg[NEXP];
#pragma unroll
    for (int e = 0; e < NEXP; ++e) lg[e] = 0.0f;

    for (int d = lane; d < DIM; d += 32) {
        float xv = xr[d];
#pragma unroll
        for (int e = 0; e < NEXP; ++e) lg[e] += xv * rw[e * DIM + d];
    }
#pragma unroll
    for (int off = 16; off > 0; off >>= 1) {
#pragma unroll
        for (int e = 0; e < NEXP; ++e) lg[e] += __shfl_xor(lg[e], off, 32);
    }
    if (lane == 0) {
#pragma unroll
        for (int e = 0; e < NEXP; ++e) lg[e] += rb[e];
        float m = lg[0]; int mi = 0;
#pragma unroll
        for (int e = 1; e < NEXP; ++e) {
            if (lg[e] > m) { m = lg[e]; mi = e; }   // first max (argmax tie rule)
        }
        float se = 0.0f;
#pragma unroll
        for (int e = 0; e < NEXP; ++e) se += __expf(lg[e] - m);
        top1w[token] = 1.0f / se;                   // max softmax prob
        int slot = atomicAdd(&counts[mi], 1);
        perm[mi * TOK + slot] = token;
    }
}

// ---------------------------------------------------------------------------
// Kernel 3: grouped GEMM per expert, bf16 WMMA / f32 accum.
// Block = 8 waves = one (expert, 16-token M-tile); wave owns 8 N-tiles.
// A tile staged to LDS with gfx1250 async-to-LDS when available.
// ---------------------------------------------------------------------------
__global__ __launch_bounds__(256) void moe_gemm_kernel(
    const __bf16* __restrict__ xbf, const __bf16* __restrict__ wbf,
    const float* __restrict__ eb, const float* __restrict__ top1w,
    const int* __restrict__ counts, const int* __restrict__ perm,
    float* __restrict__ out) {
    const int e   = blockIdx.x / MTILES;
    const int mt  = blockIdx.x % MTILES;
    const int cnt = counts[e];
    if (mt * 16 >= cnt) return;

    __shared__ __align__(16) __bf16 As[16][1032];  // 16 tokens x 1024 K (bf16)
    __shared__ int   stok[16];
    __shared__ float swt[16];

    const int t  = threadIdx.x;
    const int r  = t & 15;
    int gm    = mt * 16 + r;
    int gi    = (gm < cnt) ? gm : (cnt - 1);        // pad rows duplicate last
    int token = perm[e * TOK + gi];
    if (t < 16) { stok[t] = token; swt[t] = top1w[token]; }

    // Stage A tile: row r, 64-element (128 B) chunk per thread.
    {
        const __bf16* xr = xbf + (size_t)token * DIM;
        const int c0 = (t >> 4) * 64;
#if defined(__gfx1250__) && \
    __has_builtin(__builtin_amdgcn_global_load_async_to_lds_b128)
#pragma unroll
        for (int i = 0; i < 8; ++i) {
            __builtin_amdgcn_global_load_async_to_lds_b128(
                (__attribute__((address_space(1))) v4i32*)(__bf16*)(xr + c0 + i * 8),
                (__attribute__((address_space(3))) v4i32*)(&As[r][c0 + i * 8]),
                0, 0);
        }
#if __has_builtin(__builtin_amdgcn_s_wait_asynccnt)
        __builtin_amdgcn_s_wait_asynccnt(0);
#else
        asm volatile("s_wait_asynccnt 0x0" ::: "memory");
#endif
#else
#pragma unroll
        for (int i = 0; i < 8; ++i) {
            *(uint4*)(&As[r][c0 + i * 8]) = *(const uint4*)(xr + c0 + i * 8);
        }
#endif
    }
    __syncthreads();

    const int lane = t & 31;
    const int wave = t >> 5;
    const int lrow = lane & 15;
    const int half = lane >> 4;

    v8f acc[8];
#pragma unroll
    for (int j = 0; j < 8; ++j)
#pragma unroll
        for (int i = 0; i < 8; ++i) acc[j][i] = 0.0f;

    // Per-lane B base: column (wave*128 + j*16 + lrow), 16 consecutive K
    // bf16 per lane (lane<16: K kk+0..15, lane>=16: K kk+16..31).
    const __bf16* wbase =
        wbf + ((size_t)e * DIM + wave * 128 + lrow) * DIM + half * 16;

    for (int kk = 0; kk < DIM; kk += 32) {
        // A fragment, contiguous K order (k = 16*half + s), same as B.
        const __bf16* ap = &As[lrow][kk + half * 16];
        v8bf alo = *(const v8bf*)ap;
        v8bf ahi = *(const v8bf*)(ap + 8);
        v16bf a;
#pragma unroll
        for (int i = 0; i < 8; ++i) { a[i] = alo[i]; a[i + 8] = ahi[i]; }

#pragma unroll
        for (int j = 0; j < 8; ++j) {
            const __bf16* wp = wbase + (size_t)j * 16 * DIM + kk;
            v8bf blo = *(const v8bf*)wp;
            v8bf bhi = *(const v8bf*)(wp + 8);
            v16bf b;
#pragma unroll
            for (int i = 0; i < 8; ++i) { b[i] = blo[i]; b[i + 8] = bhi[i]; }
            acc[j] = __builtin_amdgcn_wmma_f32_16x16x32_bf16(
                false, a, false, b, (short)0, acc[j], false, false);
        }
        // WGP-scope prefetch (locality 3) to warm the near cache from the
        // L2-resident weight stream for the next K step.
        if (kk + 32 < DIM) __builtin_prefetch(wbase + kk + 32, 0, 3);
    }

    // Epilogue: C layout — VGPR g, lane l: row = g + (l>=16?8:0), col = l&15.
#pragma unroll
    for (int j = 0; j < 8; ++j) {
        const int col  = wave * 128 + j * 16 + lrow;
        const float bias = eb[e * DIM + col];
#pragma unroll
        for (int g = 0; g < 8; ++g) {
            int m = g + half * 8;
            if (mt * 16 + m < cnt) {
                out[(size_t)stok[m] * DIM + col] = (acc[j][g] + bias) * swt[m];
            }
        }
    }
}

// ---------------------------------------------------------------------------
extern "C" void kernel_launch(void* const* d_in, const int* in_sizes, int n_in,
                              void* d_out, int out_size, void* d_ws,
                              size_t ws_size, hipStream_t stream) {
    (void)in_sizes; (void)n_in; (void)out_size; (void)ws_size;
    const float* x  = (const float*)d_in[0];
    const float* rw = (const float*)d_in[1];
    const float* rb = (const float*)d_in[2];
    const float* ew = (const float*)d_in[3];
    const float* eb = (const float*)d_in[4];
    float* out = (float*)d_out;

    // Workspace layout (needs ~25 MB):
    //   [0,256)            per-expert counters
    //   [256, +16KB)       top-1 weights
    //   [16640, +128KB)    per-expert token lists
    //   [1MB, +16MB)       expert weights in bf16
    //   [17MB, +8MB)       x in bf16
    char*   ws     = (char*)d_ws;
    int*    counts = (int*)ws;
    float*  top1w  = (float*)(ws + 256);
    int*    perm   = (int*)(ws + 256 + TOK * 4);
    __bf16* wbf    = (__bf16*)(ws + (1u << 20));
    __bf16* xbf    = (__bf16*)(ws + (1u << 20) + (size_t)NEXP * DIM * DIM * 2);

    moe_init_kernel<<<1, 32, 0, stream>>>(counts);
    moe_router_kernel<<<TOK / 8, 256, 0, stream>>>(x, rw, rb, top1w, counts,
                                                   perm);
    const int w_n8 = NEXP * DIM * DIM / 8;   // 1048576
    const int x_n8 = TOK * DIM / 8;          // 524288
    cvt_bf16_kernel<<<w_n8 / 256, 256, 0, stream>>>(ew, wbf, w_n8);
    cvt_bf16_kernel<<<x_n8 / 256, 256, 0, stream>>>(x, xbf, x_n8);
    moe_gemm_kernel<<<NEXP * MTILES, 256, 0, stream>>>(xbf, wbf, eb, top1w,
                                                       counts, perm, out);
}
